// SCANDecoder_46703474376752
// MI455X (gfx1250) — matrix operand order repaired and checked by reading, not compiled, not run on hardware
//
#include <hip/hip_runtime.h>
#include <cstdint>
#include <cstddef>

typedef __bf16 bf16;
typedef bf16  v16bf  __attribute__((ext_vector_type(16)));
typedef bf16  bf16x8 __attribute__((ext_vector_type(8)));
typedef float v8f    __attribute__((ext_vector_type(8)));

#define B_  256
#define T_  128
#define S_  256
#define D_  512
#define V_  50
#define H_  4
#define HD_ 128
#define ROWS 16   // batch rows per workgroup (WMMA M)

// ---------------- WMMA helpers (CDNA5 wave32, 16x16x32 bf16) ----------------

__device__ __forceinline__ v8f wmma_bf(v16bf a, v16bf b, v8f c) {
  // D = A(16x32) * B(32x16) + C, f32 accumulate
  return __builtin_amdgcn_wmma_f32_16x16x32_bf16(false, a, false, b, (short)0, c,
                                                 false, false);
}

// A fragment: 16x32 bf16 tile from row-major bf16 source (LDS), leading dim ld.
// Layout used here (contiguous k, not ISA table order): lane L: M=L%16, g=L/16;
//   elems 0..15 -> K = k0 + 16g + e       (contiguous 32B, matches packed B)
__device__ __forceinline__ v16bf a_frag_bf16(const bf16* src, int ld, int k0) {
  const int lane = threadIdx.x & 31;
  const int m = lane & 15, g = lane >> 4;
  const bf16* row = src + m * ld;
  v16bf a;
#pragma unroll
  for (int e = 0; e < 8; ++e) a[e] = row[k0 + 16 * g + e];
#pragma unroll
  for (int e = 0; e < 8; ++e) a[8 + e] = row[k0 + 16 * g + 8 + e];
  return a;
}

// B fragment from PACKED weights: tile (kt,nt) of a [K][N] matrix is stored as
// 512 contiguous bf16, ordered [lane][elem] to match the A fragment's contiguous-k order:
//   lane L -> column n = nt*16 + L%16 ; elem e -> K = kt*32 + 16*(L/16) + e
// so each lane does ONE aligned 32-byte vector load; a wave reads 1KB contiguous.
__device__ __forceinline__ v16bf b_frag_pk(const bf16* P, int ktiles, int kt,
                                           int nt) {
  const int lane = threadIdx.x & 31;
  const bf16* p = P + (((size_t)nt * ktiles + kt) << 9) + lane * 16;
  return *(const v16bf*)p;
}

// C/D layout: lanes 0-15: vgpr i -> (M=i, N=lane); lanes 16-31: (M=i+8, N=lane-16)

// ---------------- Kernel 1: weight conversion into packed WMMA-B order -----

// src is [Nd_valid][Kd] row-major f32 (out_features x in_features).
// dst is packed bf16, Kd x Nd (Nd >= Nd_valid, zero padded).
__device__ __forceinline__ void pack_b(bf16* dst, const float* src, int Kd,
                                       int Nd, int Nvalid, size_t i0,
                                       size_t stride) {
  const int ktiles = Kd >> 5;
  const size_t total = (size_t)Kd * Nd;
  for (size_t i = i0; i < total; i += stride) {
    const int within = (int)(i & 511);
    const size_t tile = i >> 9;
    const int lane = within >> 4, e = within & 15;
    const int kt = (int)(tile % ktiles), nt = (int)(tile / ktiles);
    const int k = kt * 32 + ((lane >> 4) << 4) + e;
    const int n = nt * 16 + (lane & 15);
    dst[i] = (n < Nvalid) ? (bf16)src[(size_t)n * Kd + k] : (bf16)0.0f;
  }
}

__global__ void prep_weights_kernel(const float* __restrict__ in_proj_w,
                                    const float* __restrict__ out_proj_w,
                                    const float* __restrict__ gru_w_ih,
                                    const float* __restrict__ gru_w_hh,
                                    const float* __restrict__ out_w,
                                    bf16* Wq_pk, bf16* Wk_pk, bf16* Wv_pk,
                                    bf16* op_pk, bf16* gih_pk, bf16* ghh_pk,
                                    bf16* ow_pk) {
  const size_t tid0 = (size_t)blockIdx.x * blockDim.x + threadIdx.x;
  const size_t stride = (size_t)gridDim.x * blockDim.x;
  pack_b(Wq_pk, in_proj_w, D_, D_, D_, tid0, stride);
  pack_b(Wk_pk, in_proj_w + (size_t)D_ * D_, D_, D_, D_, tid0, stride);
  pack_b(Wv_pk, in_proj_w + (size_t)2 * D_ * D_, D_, D_, D_, tid0, stride);
  pack_b(op_pk, out_proj_w, D_, D_, D_, tid0, stride);
  pack_b(gih_pk, gru_w_ih, 2 * D_, 3 * D_, 3 * D_, tid0, stride);
  pack_b(ghh_pk, gru_w_hh, D_, 3 * D_, 3 * D_, tid0, stride);
  pack_b(ow_pk, out_w, D_, 128, V_, tid0, stride);
}

// ---------------- Kernel 2: K/V projection (WMMA GEMM, 65536x512x512 x2) ----

__global__ __launch_bounds__(256) void kv_proj_kernel(
    const float* __restrict__ enc_seq, const float* __restrict__ in_proj_b,
    const bf16* __restrict__ Wk_pk, const bf16* __restrict__ Wv_pk,
    bf16* __restrict__ Khb, bf16* __restrict__ Vhb) {
  __shared__ bf16 at[ROWS * D_];  // 16 KB staged A tile
  const int tid = threadIdx.x, lane = tid & 31, wave = tid >> 5;
  const size_t m0 = (size_t)blockIdx.x * ROWS;  // global rows of (B*S, D)

  for (int i = tid; i < ROWS * D_; i += 256) {
    const int r = i >> 9, j = i & (D_ - 1);
    at[i] = (bf16)enc_seq[(m0 + r) * D_ + j];
  }
  __syncthreads();

#pragma unroll 1
  for (int which = 0; which < 2; ++which) {
    const bf16* W = which ? Wv_pk : Wk_pk;
    const float* bias = in_proj_b + (which ? 2 * D_ : D_);
    bf16* dst = which ? Vhb : Khb;
    for (int nt = wave; nt < D_ / 16; nt += 8) {
      v8f acc = {0.f, 0.f, 0.f, 0.f, 0.f, 0.f, 0.f, 0.f};
#pragma unroll 4
      for (int kt = 0; kt < D_ / 32; ++kt) {
        if (kt + 1 < D_ / 32)
          __builtin_prefetch(W + (((size_t)nt * (D_ / 32) + kt + 1) << 9), 0, 3);
        acc = wmma_bf(a_frag_bf16(at, D_, kt * 32),
                      b_frag_pk(W, D_ / 32, kt, nt), acc);
      }
      const int n = nt * 16 + (lane & 15), g = lane >> 4;
      const int h = n >> 7, d = n & (HD_ - 1);
      const float bv = bias[n];
#pragma unroll
      for (int i = 0; i < 8; ++i) {
        const size_t row = m0 + 8 * g + i;
        const int b = (int)(row >> 8), s = (int)(row & (S_ - 1));
        dst[(((size_t)b * H_ + h) * S_ + s) * HD_ + d] = (bf16)(acc[i] + bv);
      }
    }
  }
}

// ---------------- Kernel 3: persistent sequential decoder ------------------
// grid = B/16 workgroups; each WG owns 16 batch rows through all 127 steps.

__global__ __launch_bounds__(256, 1) void decode_kernel(
    const float* __restrict__ init_hidden, const int* __restrict__ actions,
    const unsigned char* __restrict__ enc_mask,
    const float* __restrict__ emb_table, const float* __restrict__ in_proj_b,
    const float* __restrict__ out_proj_b, const float* __restrict__ gru_b_ih,
    const float* __restrict__ gru_b_hh, const float* __restrict__ ln_gamma,
    const float* __restrict__ ln_beta, const float* __restrict__ out_b,
    const bf16* __restrict__ Wq_pk, const bf16* __restrict__ op_pk,
    const bf16* __restrict__ gih_pk, const bf16* __restrict__ ghh_pk,
    const bf16* __restrict__ ow_pk, const bf16* __restrict__ Khb,
    const bf16* __restrict__ Vhb, float* __restrict__ out) {
  __shared__ float hid[ROWS * D_];     // 32 KB  persistent hidden (f32, exact)
  __shared__ bf16 hbf[ROWS * D_];      // 16 KB  bf16 mirror (WMMA A operand)
  __shared__ float qxg[ROWS * D_];     // 32 KB  q, then xg gate block
  __shared__ float hgb[ROWS * D_];     // 32 KB  hg gate block
  __shared__ bf16 ctxp[ROWS * D_];     // 16 KB  attention ctx (pre out_proj)
  __shared__ bf16 gib[ROWS * 2 * D_];  // 32 KB  gi=[emb|ctx]; reused for normed
  // total 160 KB LDS / workgroup (<= 320 KB WGP budget)

  const int tid = threadIdx.x, lane = tid & 31, wave = tid >> 5;
  const int b0 = blockIdx.x * ROWS;
  const float scale = 0.08838834764831845f;  // 1/sqrt(128)

  for (int i = tid; i < ROWS * D_; i += 256) {
    const int r = i >> 9, j = i & (D_ - 1);
    const float v = init_hidden[(size_t)(b0 + r) * D_ + j];
    hid[i] = v;
    hbf[i] = (bf16)v;
  }
  __syncthreads();

  float rreg[32], zreg[32];

#pragma unroll 1
  for (int t = 0; t < T_ - 1; ++t) {
    // ---- embedding gather into gi[:, 0:D) ----
    for (int i = tid; i < ROWS * D_; i += 256) {
      const int r = i >> 9, j = i & (D_ - 1);
      const int act = actions[(b0 + r) * T_ + t];
      gib[r * (2 * D_) + j] = (bf16)emb_table[(size_t)act * D_ + j];
    }
    // ---- q = (hidden @ WqT + bq) * scale  (WMMA 16x512x512) ----
    for (int nt = wave; nt < D_ / 16; nt += 8) {
      v8f acc = {0.f, 0.f, 0.f, 0.f, 0.f, 0.f, 0.f, 0.f};
#pragma unroll 4
      for (int kt = 0; kt < D_ / 32; ++kt) {
        if (kt + 1 < D_ / 32)
          __builtin_prefetch(Wq_pk + (((size_t)nt * (D_ / 32) + kt + 1) << 9), 0, 3);
        acc = wmma_bf(a_frag_bf16(hbf, D_, kt * 32),
                      b_frag_pk(Wq_pk, D_ / 32, kt, nt), acc);
      }
      const int n = nt * 16 + (lane & 15), g = lane >> 4;
      const float bias = in_proj_b[n];
#pragma unroll
      for (int i = 0; i < 8; ++i) qxg[(8 * g + i) * D_ + n] = (acc[i] + bias) * scale;
    }
    __syncthreads();

    // ---- attention: wave w owns batch rows 2w, 2w+1 (no cross-wave deps) ----
#pragma unroll 1
    for (int rr = 0; rr < 2; ++rr) {
      const int r = wave * 2 + rr, b = b0 + r;
#pragma unroll 1
      for (int h = 0; h < H_; ++h) {
        const float* qrow = &qxg[r * D_ + h * HD_];
        const bf16* Kb = Khb + ((size_t)b * H_ + h) * S_ * HD_;
        float svals[8];
        float smax = -3.4028235e38f;
#pragma unroll
        for (int c = 0; c < 8; ++c) {
          const int s = lane + c * 32;
          const bf16* kr = Kb + (size_t)s * HD_;
          float dot = 0.f;
#pragma unroll
          for (int d = 0; d < HD_; d += 8) {
            const bf16x8 kv = *(const bf16x8*)(kr + d);
#pragma unroll
            for (int e = 0; e < 8; ++e) dot += qrow[d + e] * (float)kv[e];
          }
          if (!enc_mask[b * S_ + s]) dot = -3.4028235e38f;
          svals[c] = dot;
          smax = fmaxf(smax, dot);
        }
#pragma unroll
        for (int m = 16; m >= 1; m >>= 1) smax = fmaxf(smax, __shfl_xor(smax, m, 32));
        float ssum = 0.f;
#pragma unroll
        for (int c = 0; c < 8; ++c) {
          const float e = __expf(svals[c] - smax);
          svals[c] = e;
          ssum += e;
        }
#pragma unroll
        for (int m = 16; m >= 1; m >>= 1) ssum += __shfl_xor(ssum, m, 32);
        const float inv = 1.f / fmaxf(ssum, 1e-30f);
        // ctx accumulation: broadcast attn via shuffle, coalesced V reads
        float cacc[4] = {0.f, 0.f, 0.f, 0.f};
        const bf16* Vb = Vhb + ((size_t)b * H_ + h) * S_ * HD_;
#pragma unroll
        for (int c = 0; c < 8; ++c) {
          const float av = svals[c] * inv;
          for (int sl = 0; sl < 32; ++sl) {
            const float a = __shfl(av, sl, 32);
            const bf16* vr = Vb + (size_t)(c * 32 + sl) * HD_;
#pragma unroll
            for (int dj = 0; dj < 4; ++dj)
              cacc[dj] += a * (float)vr[lane + dj * 32];
          }
        }
#pragma unroll
        for (int dj = 0; dj < 4; ++dj)
          ctxp[r * D_ + h * HD_ + lane + dj * 32] = (bf16)cacc[dj];
      }
    }
    __syncthreads();

    // ---- ctx @ out_projT + b -> gi[:, D:2D)  (WMMA 16x512x512) ----
    for (int nt = wave; nt < D_ / 16; nt += 8) {
      v8f acc = {0.f, 0.f, 0.f, 0.f, 0.f, 0.f, 0.f, 0.f};
#pragma unroll 4
      for (int kt = 0; kt < D_ / 32; ++kt) {
        if (kt + 1 < D_ / 32)
          __builtin_prefetch(op_pk + (((size_t)nt * (D_ / 32) + kt + 1) << 9), 0, 3);
        acc = wmma_bf(a_frag_bf16(ctxp, D_, kt * 32),
                      b_frag_pk(op_pk, D_ / 32, kt, nt), acc);
      }
      const int n = nt * 16 + (lane & 15), g = lane >> 4;
      const float bias = out_proj_b[n];
#pragma unroll
      for (int i = 0; i < 8; ++i)
        gib[(8 * g + i) * (2 * D_) + D_ + n] = (bf16)(acc[i] + bias);
    }
    __syncthreads();

    // ---- GRU gates in 3 blocks of 512: r, z, n ----
#pragma unroll 1
    for (int gb = 0; gb < 3; ++gb) {
      // xg block = gi(16x1024) @ gihT[:, gb*512 + n)   (WMMA 16x512x1024)
      for (int nt = wave; nt < D_ / 16; nt += 8) {
        const int gnt = gb * 32 + nt;
        v8f acc = {0.f, 0.f, 0.f, 0.f, 0.f, 0.f, 0.f, 0.f};
#pragma unroll 4
        for (int kt = 0; kt < (2 * D_) / 32; ++kt) {
          if (kt + 1 < (2 * D_) / 32)
            __builtin_prefetch(gih_pk + (((size_t)gnt * 32 + kt + 1) << 9), 0, 3);
          acc = wmma_bf(a_frag_bf16(gib, 2 * D_, kt * 32),
                        b_frag_pk(gih_pk, 32, kt, gnt), acc);
        }
        const int n = nt * 16 + (lane & 15), g = lane >> 4;
#pragma unroll
        for (int i = 0; i < 8; ++i) qxg[(8 * g + i) * D_ + n] = acc[i];
      }
      // hg block = hidden(16x512) @ ghhT[:, gb*512 + n) (WMMA 16x512x512)
      for (int nt = wave; nt < D_ / 16; nt += 8) {
        const int gnt = gb * 32 + nt;
        v8f acc = {0.f, 0.f, 0.f, 0.f, 0.f, 0.f, 0.f, 0.f};
#pragma unroll 4
        for (int kt = 0; kt < D_ / 32; ++kt) {
          if (kt + 1 < D_ / 32)
            __builtin_prefetch(ghh_pk + (((size_t)gnt * 16 + kt + 1) << 9), 0, 3);
          acc = wmma_bf(a_frag_bf16(hbf, D_, kt * 32),
                        b_frag_pk(ghh_pk, 16, kt, gnt), acc);
        }
        const int n = nt * 16 + (lane & 15), g = lane >> 4;
#pragma unroll
        for (int i = 0; i < 8; ++i) hgb[(8 * g + i) * D_ + n] = acc[i];
      }
      __syncthreads();
      // elementwise: thread owns fixed (r, j) slice across all gate blocks
      {
        const int r = tid >> 4, jb = (tid & 15) * 32;
#pragma unroll
        for (int c = 0; c < 32; ++c) {
          const int j = jb + c;
          const float x = qxg[r * D_ + j] + gru_b_ih[gb * D_ + j];
          const float hg = hgb[r * D_ + j] + gru_b_hh[gb * D_ + j];
          if (gb == 0) {
            rreg[c] = 1.f / (1.f + __expf(-(x + hg)));
          } else if (gb == 1) {
            zreg[c] = 1.f / (1.f + __expf(-(x + hg)));
          } else {
            const float nv = tanhf(x + rreg[c] * hg);
            const float hold = hid[r * D_ + j];
            const float hnew = (1.f - zreg[c]) * nv + zreg[c] * hold;
            hid[r * D_ + j] = hnew;
            hbf[r * D_ + j] = (bf16)hnew;
          }
        }
      }
      __syncthreads();
    }

    // ---- LayerNorm: wave w handles rows 2w, 2w+1; normed -> gi[:, 0:D) ----
#pragma unroll 1
    for (int rr = 0; rr < 2; ++rr) {
      const int r = wave * 2 + rr;
      float vals[16];
      float s1 = 0.f;
#pragma unroll
      for (int c = 0; c < 16; ++c) {
        const float v = hid[r * D_ + lane + c * 32];
        vals[c] = v;
        s1 += v;
      }
#pragma unroll
      for (int m = 16; m >= 1; m >>= 1) s1 += __shfl_xor(s1, m, 32);
      const float mu = s1 * (1.f / D_);
      float s2 = 0.f;
#pragma unroll
      for (int c = 0; c < 16; ++c) {
        const float dv = vals[c] - mu;
        s2 += dv * dv;
      }
#pragma unroll
      for (int m = 16; m >= 1; m >>= 1) s2 += __shfl_xor(s2, m, 32);
      const float rstd = rsqrtf(s2 * (1.f / D_) + 1e-5f);
#pragma unroll
      for (int c = 0; c < 16; ++c) {
        const int d = lane + c * 32;
        const float nv = (vals[c] - mu) * rstd * ln_gamma[d] + ln_beta[d];
        gib[r * (2 * D_) + d] = (bf16)nv;
      }
    }
    __syncthreads();

    // ---- logits = normed @ owT + out_b  (WMMA 16x128x512, N padded to 128) ----
    {
      const int nt = wave;  // 8 waves cover padded N=128
      v8f acc = {0.f, 0.f, 0.f, 0.f, 0.f, 0.f, 0.f, 0.f};
#pragma unroll 4
      for (int kt = 0; kt < D_ / 32; ++kt)
        acc = wmma_bf(a_frag_bf16(gib, 2 * D_, kt * 32),
                      b_frag_pk(ow_pk, 16, kt, nt), acc);
      const int n = nt * 16 + (lane & 15), g = lane >> 4;
      if (n < V_) {
        const float bias = out_b[n];
#pragma unroll
        for (int i = 0; i < 8; ++i) {
          const int b = b0 + 8 * g + i;
          out[((size_t)b * (T_ - 1) + t) * V_ + n] = acc[i] + bias;
        }
      }
    }
    __syncthreads();
  }
}

// ---------------- host side ------------------------------------------------

extern "C" void kernel_launch(void* const* d_in, const int* in_sizes, int n_in,
                              void* d_out, int out_size, void* d_ws, size_t ws_size,
                              hipStream_t stream) {
  (void)in_sizes; (void)n_in; (void)out_size; (void)ws_size;
  const float* init_hidden = (const float*)d_in[0];
  const int* target_actions = (const int*)d_in[1];
  const float* enc_seq = (const float*)d_in[2];
  const unsigned char* enc_mask = (const unsigned char*)d_in[3];
  const float* emb_table = (const float*)d_in[4];
  const float* in_proj_w = (const float*)d_in[5];
  const float* in_proj_b = (const float*)d_in[6];
  const float* out_proj_w = (const float*)d_in[7];
  const float* out_proj_b = (const float*)d_in[8];
  const float* gru_w_ih = (const float*)d_in[9];
  const float* gru_w_hh = (const float*)d_in[10];
  const float* gru_b_ih = (const float*)d_in[11];
  const float* gru_b_hh = (const float*)d_in[12];
  const float* ln_gamma = (const float*)d_in[13];
  const float* ln_beta = (const float*)d_in[14];
  const float* out_w = (const float*)d_in[15];
  const float* out_b = (const float*)d_in[16];

  char* ws = (char*)d_ws;
  size_t off = 0;
  auto take = [&](size_t bytes) {
    char* p = ws + off;
    off += (bytes + 255) & ~(size_t)255;
    return p;
  };
  bf16* Wq_pk = (bf16*)take((size_t)D_ * D_ * 2);
  bf16* Wk_pk = (bf16*)take((size_t)D_ * D_ * 2);
  bf16* Wv_pk = (bf16*)take((size_t)D_ * D_ * 2);
  bf16* op_pk = (bf16*)take((size_t)D_ * D_ * 2);
  bf16* gih_pk = (bf16*)take((size_t)(2 * D_) * (3 * D_) * 2);
  bf16* ghh_pk = (bf16*)take((size_t)D_ * (3 * D_) * 2);
  bf16* ow_pk = (bf16*)take((size_t)D_ * 128 * 2);
  bf16* Khb = (bf16*)take((size_t)B_ * H_ * S_ * HD_ * 2);
  bf16* Vhb = (bf16*)take((size_t)B_ * H_ * S_ * HD_ * 2);

  prep_weights_kernel<<<1024, 256, 0, stream>>>(in_proj_w, out_proj_w, gru_w_ih,
                                                gru_w_hh, out_w, Wq_pk, Wk_pk,
                                                Wv_pk, op_pk, gih_pk, ghh_pk,
                                                ow_pk);
  kv_proj_kernel<<<(B_ * S_) / ROWS, 256, 0, stream>>>(enc_seq, in_proj_b,
                                                       Wk_pk, Wv_pk, Khb, Vhb);
  decode_kernel<<<B_ / ROWS, 256, 0, stream>>>(
      init_hidden, target_actions, enc_mask, emb_table, in_proj_b, out_proj_b,
      gru_b_ih, gru_b_hh, ln_gamma, ln_beta, out_b, Wq_pk, op_pk, gih_pk,
      ghh_pk, ow_pk, Khb, Vhb, (float*)d_out);
}
